// Attention_57389353009692
// MI455X (gfx1250) — matrix operand order repaired and checked by reading, not compiled, not run on hardware
//
#include <hip/hip_runtime.h>
#include <stdint.h>

#define S_LEN   2048
#define D_MODEL 2048
#define N_HEADS 32
#define N_KVH   8
#define HEAD_DIM 64
#define NKV     (N_KVH * HEAD_DIM)   // 512
#define N_EXP   8
#define LORA_R  8
#define LORA_SCALE 4.0f

typedef __bf16 bf16_t;
typedef __attribute__((ext_vector_type(16))) __bf16 v16bf;
typedef __attribute__((ext_vector_type(8)))  __bf16 v8bf;
typedef __attribute__((ext_vector_type(8)))  float  v8f;

// ---------------------------------------------------------------------------
// WMMA helper: D = A(16x32 bf16) * B(32x16 bf16) + C(16x16 f32)
// ---------------------------------------------------------------------------
__device__ __forceinline__ v8f wmma_bf16(v16bf a, v16bf b, v8f c) {
    return __builtin_amdgcn_wmma_f32_16x16x32_bf16(
        /*neg_a=*/false, a, /*neg_b=*/false, b,
        /*c_mod=*/(short)0, c, /*reuse_a=*/false, /*reuse_b=*/false);
}

// A-fragment (16x32, 16-bit): lane = (half<<4)|m ; elements 0..7 <-> K=half*8+0..7,
// elements 8..15 <-> K=16+half*8+0..7.  Caller passes p = base + m*stride + half*8.
__device__ __forceinline__ v16bf frag_a(const bf16_t* p) {
    v8bf x = *(const v8bf*)p;
    v8bf y = *(const v8bf*)(p + 16);
    return __builtin_shufflevector(x, y, 0,1,2,3,4,5,6,7,8,9,10,11,12,13,14,15);
}

// B-fragment (32x16, 16-bit): lane = (half<<4)|n ; elements 0..7 <-> K=half*8+0..7,
// elements 8..15 <-> K=16+half*8+0..7.  Caller passes p = base + n*stride + half*8.
__device__ __forceinline__ v16bf frag_b(const bf16_t* p) {
    v8bf x = *(const v8bf*)p;
    v8bf y = *(const v8bf*)(p + 16);
    return __builtin_shufflevector(x, y, 0,1,2,3,4,5,6,7,8,9,10,11,12,13,14,15);
}

// ---------------------------------------------------------------------------
// CDNA5 async global -> LDS copy (ASYNCcnt-tracked, bypasses VGPR round-trip).
// lds_off = raw LDS byte offset (low 32 bits of a generic shared-mem address:
// LDS aperture is 4GB-aligned).
// ---------------------------------------------------------------------------
__device__ __forceinline__ void async_ld16(uint32_t lds_off, const void* g) {
    asm volatile("global_load_async_to_lds_b128 %0, %1, off"
                 :: "v"(lds_off), "v"(g) : "memory");
}
__device__ __forceinline__ void wait_async0() {
    asm volatile("s_wait_asynccnt 0" ::: "memory");
}
__device__ __forceinline__ uint32_t lds_off_of(const void* p) {
    return (uint32_t)(uintptr_t)p;
}

// ---------------------------------------------------------------------------
// fp32 -> bf16 hi/lo split (3xBF16 emulation of fp32 GEMM)
// ---------------------------------------------------------------------------
__global__ void split_bf16_kernel(const float* __restrict__ x,
                                  bf16_t* __restrict__ hi,
                                  bf16_t* __restrict__ lo, int n) {
    for (int i = blockIdx.x * blockDim.x + threadIdx.x; i < n;
         i += gridDim.x * blockDim.x) {
        float v = x[i];
        bf16_t h = (bf16_t)v;
        hi[i] = h;
        lo[i] = (bf16_t)(v - (float)h);
    }
}

// ---------------------------------------------------------------------------
// MoE-LoRA:  out[s,o] = SCALE * sum_{e,r} rw[s,e] * (x[s,:].A[e,r,:]) * B[e,o,r]
// One 256-thread block (8 waves) per sequence row.
// ---------------------------------------------------------------------------
__global__ __launch_bounds__(256)
void lora_kernel(const float* __restrict__ x, const float* __restrict__ router,
                 const float* __restrict__ A, const float* __restrict__ Bw,
                 float* __restrict__ out, int out_dim) {
    __shared__ float xs[D_MODEL];
    __shared__ float lg[N_EXP];
    __shared__ float rw[N_EXP];
    __shared__ float hc[N_EXP * LORA_R];

    const int s = blockIdx.x;
    const int tid = threadIdx.x, lane = tid & 31, wid = tid >> 5;
    const float* xr = x + (size_t)s * D_MODEL;
    for (int i = tid; i < D_MODEL; i += 256) xs[i] = xr[i];
    __syncthreads();

    // router logits: wave w -> expert w
    {
        const float* rp = router + (size_t)wid * D_MODEL;
        float acc = 0.f;
        for (int i = lane; i < D_MODEL; i += 32) acc += xs[i] * rp[i];
        for (int m = 16; m; m >>= 1) acc += __shfl_xor(acc, m, 32);
        if (lane == 0) lg[wid] = acc;
    }
    __syncthreads();
    if (tid == 0) {
        float mx = lg[0];
        for (int e = 1; e < N_EXP; ++e) mx = fmaxf(mx, lg[e]);
        float sum = 0.f;
        for (int e = 0; e < N_EXP; ++e) { float t = __expf(lg[e] - mx); rw[e] = t; sum += t; }
        float inv = 1.f / sum;
        for (int e = 0; e < N_EXP; ++e) rw[e] *= inv;
    }
    __syncthreads();

    // h coefficients: wave w handles expert e = w, all ranks r
    for (int r = 0; r < LORA_R; ++r) {
        const float* ap = A + ((size_t)wid * LORA_R + r) * D_MODEL;
        float acc = 0.f;
        for (int i = lane; i < D_MODEL; i += 32) acc += xs[i] * ap[i];
        for (int m = 16; m; m >>= 1) acc += __shfl_xor(acc, m, 32);
        if (lane == 0) hc[wid * LORA_R + r] = LORA_SCALE * rw[wid] * acc;
    }
    __syncthreads();

    for (int o = tid; o < out_dim; o += 256) {
        float acc = 0.f;
        #pragma unroll
        for (int e = 0; e < N_EXP; ++e) {
            const float* bp = Bw + ((size_t)e * out_dim + o) * LORA_R;
            #pragma unroll
            for (int r = 0; r < LORA_R; ++r) acc += hc[e * LORA_R + r] * bp[r];
        }
        out[(size_t)s * out_dim + o] = acc;
    }
}

// ---------------------------------------------------------------------------
// Split-bf16 WMMA GEMM:  Y[M,N] = A[M,K] * W[N,K]^T + bias
// Double-buffered LDS, async global->LDS staging (ASYNCcnt), one barrier/step.
// Block tile 128x128, K-step 32, 8 waves (2M x 4N), each wave 64x32 output.
// ---------------------------------------------------------------------------
__global__ __launch_bounds__(256)
void gemm_split_kernel(const bf16_t* __restrict__ Ah, const bf16_t* __restrict__ Al,
                       const bf16_t* __restrict__ Wh, const bf16_t* __restrict__ Wl,
                       const float* __restrict__ bias, float* __restrict__ Y,
                       int N, int K) {
    const int LDT  = 40;          // 32 + 8 pad, keeps 16B alignment
    const int TILE = 128 * LDT;   // elems per (buf, array)
    __shared__ __align__(16) bf16_t smem[2 * 4 * 128 * 40];  // 80 KB

    const int tid = threadIdx.x, lane = tid & 31, wid = tid >> 5;
    const int wm = wid >> 2, wn = wid & 3;
    const int brow = blockIdx.y * 128, bcol = blockIdx.x * 128;
    const int fm = lane & 15, fh8 = (lane >> 4) * 8, fh16 = (lane >> 4) * 16;

    const v8f zero8 = {0.f, 0.f, 0.f, 0.f, 0.f, 0.f, 0.f, 0.f};
    v8f acc[4][2];
    #pragma unroll
    for (int mi = 0; mi < 4; ++mi)
        #pragma unroll
        for (int ni = 0; ni < 2; ++ni) acc[mi][ni] = zero8;

    const int sr = tid >> 1;        // staging row 0..127
    const int sc = (tid & 1) * 16;  // staging col 0 / 16
    const uint32_t lds0 = lds_off_of(smem);
    const uint32_t stOff = (uint32_t)((sr * LDT + sc) * 2);

    auto issue = [&](int buf, int kb) {
        uint32_t b = lds0 + (uint32_t)(buf * 4 * TILE) * 2 + stOff;
        size_t ga = (size_t)(brow + sr) * K + kb + sc;
        size_t gb = (size_t)(bcol + sr) * K + kb + sc;
        async_ld16(b + 0u * TILE * 2,      Ah + ga);
        async_ld16(b + 0u * TILE * 2 + 16, Ah + ga + 8);
        async_ld16(b + 1u * TILE * 2,      Al + ga);
        async_ld16(b + 1u * TILE * 2 + 16, Al + ga + 8);
        async_ld16(b + 2u * TILE * 2,      Wh + gb);
        async_ld16(b + 2u * TILE * 2 + 16, Wh + gb + 8);
        async_ld16(b + 3u * TILE * 2,      Wl + gb);
        async_ld16(b + 3u * TILE * 2 + 16, Wl + gb + 8);
    };

    issue(0, 0);
    int buf = 0;
    for (int kb = 0; kb < K; kb += 32) {
        wait_async0();        // current buffer's copies landed (this wave)
        __syncthreads();      // ... and everyone else's
        if (kb + 32 < K) issue(buf ^ 1, kb + 32);  // prefetch next tile

        const bf16_t* Ash = smem + (buf * 4 + 0) * TILE;
        const bf16_t* Asl = smem + (buf * 4 + 1) * TILE;
        const bf16_t* Bsh = smem + (buf * 4 + 2) * TILE;
        const bf16_t* Bsl = smem + (buf * 4 + 3) * TILE;

        v16bf ah[4], al[4], bh[2], bl[2];
        #pragma unroll
        for (int mi = 0; mi < 4; ++mi) {
            int row = wm * 64 + mi * 16 + fm;
            ah[mi] = frag_a(&Ash[row * LDT + fh8]);
            al[mi] = frag_a(&Asl[row * LDT + fh8]);
        }
        #pragma unroll
        for (int ni = 0; ni < 2; ++ni) {
            int col = wn * 32 + ni * 16 + fm;
            bh[ni] = frag_b(&Bsh[col * LDT + fh8]);
            bl[ni] = frag_b(&Bsl[col * LDT + fh8]);
        }
        #pragma unroll
        for (int mi = 0; mi < 4; ++mi)
            #pragma unroll
            for (int ni = 0; ni < 2; ++ni) {
                acc[mi][ni] = wmma_bf16(ah[mi], bh[ni], acc[mi][ni]);
                acc[mi][ni] = wmma_bf16(ah[mi], bl[ni], acc[mi][ni]);
                acc[mi][ni] = wmma_bf16(al[mi], bh[ni], acc[mi][ni]);
            }
        buf ^= 1;
    }

    // Epilogue: C/D layout -> row = 8*half + r (VGPR r), col = lane&15
    #pragma unroll
    for (int mi = 0; mi < 4; ++mi)
        #pragma unroll
        for (int ni = 0; ni < 2; ++ni)
            #pragma unroll
            for (int rr = 0; rr < 8; ++rr) {
                int row = brow + wm * 64 + mi * 16 + (lane >> 4) * 8 + rr;
                int col = bcol + wn * 32 + ni * 16 + fm;
                float v = acc[mi][ni][rr];
                if (bias) v += bias[(size_t)row * N + col];
                Y[(size_t)row * N + col] = v;
            }
}

// ---------------------------------------------------------------------------
// RoPE in-place on [S, nh, HEAD_DIM]
// ---------------------------------------------------------------------------
__global__ void rope_kernel(float* __restrict__ t, const float* __restrict__ cs,
                            const float* __restrict__ sn, int nh) {
    int idx = blockIdx.x * blockDim.x + threadIdx.x;
    int total = S_LEN * nh * (HEAD_DIM / 2);
    if (idx >= total) return;
    int i = idx & 31;        // HEAD_DIM/2 == 32
    int rem = idx >> 5;
    int h = rem % nh;
    int s = rem / nh;
    float* p = t + ((size_t)s * nh + h) * HEAD_DIM + 2 * i;
    float xr = p[0], xi = p[1];
    float c = cs[s * 32 + i], si = sn[s * 32 + i];
    p[0] = xr * c - xi * si;
    p[1] = xr * si + xi * c;
}

// ---------------------------------------------------------------------------
// Flash attention (causal): one block = (head, 128 q rows), 8 waves x 16 rows.
// Scores via split-bf16 (3 products), PV via P(bf16) x V(hi/lo) (2 products).
// K tiles staged with async global->LDS; V transposed through VGPRs.
// ---------------------------------------------------------------------------
__global__ __launch_bounds__(256)
void attn_kernel(const bf16_t* __restrict__ Qh, const bf16_t* __restrict__ Ql,
                 const bf16_t* __restrict__ Kh, const bf16_t* __restrict__ Kl,
                 const bf16_t* __restrict__ Vh, const bf16_t* __restrict__ Vl,
                 float* __restrict__ O) {
    const int LKT = 72;  // 64 + 8
    const int LVT = 40;  // 32 + 8
    __shared__ __align__(16) bf16_t Ksh[32 * 72];
    __shared__ __align__(16) bf16_t Ksl[32 * 72];
    __shared__ __align__(16) bf16_t Vth[64 * 40];   // V transposed: [hd][k]
    __shared__ __align__(16) bf16_t Vtl[64 * 40];
    __shared__ __align__(16) bf16_t Ps[8][16 * 40]; // per-wave P staging

    const int tid = threadIdx.x, lane = tid & 31, wid = tid >> 5;
    const int head = blockIdx.y, qtile = blockIdx.x;
    const int qbase = qtile * 128 + wid * 16;
    const int kvh = head >> 2;  // NREP = 4
    const int fm = lane & 15, fh8 = (lane >> 4) * 8, fh16 = (lane >> 4) * 16;

    // Q fragments (16 rows x HD=64 -> two K-chunks), hi and lo
    v16bf qfh[2], qfl[2];
    {
        size_t base = (size_t)(qbase + fm) * D_MODEL + head * HEAD_DIM;
        #pragma unroll
        for (int c = 0; c < 2; ++c) {
            qfh[c] = frag_a(&Qh[base + c * 32 + fh8]);
            qfl[c] = frag_a(&Ql[base + c * 32 + fh8]);
        }
    }

    const v8f zero8 = {0.f, 0.f, 0.f, 0.f, 0.f, 0.f, 0.f, 0.f};
    v8f acc[4];
    #pragma unroll
    for (int nf = 0; nf < 4; ++nf) acc[nf] = zero8;
    float mrow[8], lrow[8];
    #pragma unroll
    for (int rr = 0; rr < 8; ++rr) { mrow[rr] = -1e30f; lrow[rr] = 0.f; }

    const int kend = qtile * 128 + 128;  // causal: only tiles up to q block end
    const int str = tid >> 3;            // K staging row 0..31
    const int stc = (tid & 7) * 8;       // K staging col 0..56
    const uint32_t kshOff = lds_off_of(&Ksh[str * LKT + stc]);
    const uint32_t kslOff = lds_off_of(&Ksl[str * LKT + stc]);

    for (int kb = 0; kb < kend; kb += 32) {
        __syncthreads();
        {
            size_t g = (size_t)(kb + str) * NKV + kvh * HEAD_DIM + stc;
            async_ld16(kshOff, Kh + g);          // K tiles: async copy
            async_ld16(kslOff, Kl + g);
            v8bf vh8 = *(const v8bf*)&Vh[g];     // V: register transpose
            v8bf vl8 = *(const v8bf*)&Vl[g];
            #pragma unroll
            for (int i = 0; i < 8; ++i) {
                Vth[(stc + i) * LVT + str] = vh8[i];
                Vtl[(stc + i) * LVT + str] = vl8[i];
            }
        }
        wait_async0();
        __syncthreads();

        // S = Q K^T (16q x 32k), split 3-product
        v8f sf[2] = {zero8, zero8};
        #pragma unroll
        for (int nf = 0; nf < 2; ++nf)
            #pragma unroll
            for (int c = 0; c < 2; ++c) {
                v16bf kbh = frag_b(&Ksh[(nf * 16 + fm) * LKT + c * 32 + fh8]);
                v16bf kbl = frag_b(&Ksl[(nf * 16 + fm) * LKT + c * 32 + fh8]);
                sf[nf] = wmma_bf16(qfh[c], kbh, sf[nf]);
                sf[nf] = wmma_bf16(qfh[c], kbl, sf[nf]);
                sf[nf] = wmma_bf16(qfl[c], kbh, sf[nf]);
            }

        // scale + causal mask (row = 8*half + rr, col = fm within n-frag)
        const int qr0 = qbase + (lane >> 4) * 8;
        #pragma unroll
        for (int nf = 0; nf < 2; ++nf)
            #pragma unroll
            for (int rr = 0; rr < 8; ++rr) {
                float v = sf[nf][rr] * 0.125f;  // 1/sqrt(64)
                int kc = kb + nf * 16 + fm;
                sf[nf][rr] = (kc <= qr0 + rr) ? v : -1e30f;
            }

        // online softmax: row reductions across the 16 lanes of each half
        float tm[8], scl[8], ps[8];
        #pragma unroll
        for (int rr = 0; rr < 8; ++rr) tm[rr] = fmaxf(sf[0][rr], sf[1][rr]);
        #pragma unroll
        for (int m = 1; m <= 8; m <<= 1)
            #pragma unroll
            for (int rr = 0; rr < 8; ++rr)
                tm[rr] = fmaxf(tm[rr], __shfl_xor(tm[rr], m, 32));
        #pragma unroll
        for (int rr = 0; rr < 8; ++rr) {
            float mn = fmaxf(mrow[rr], tm[rr]);
            scl[rr] = __expf(mrow[rr] - mn);
            mrow[rr] = mn;
        }
        #pragma unroll
        for (int nf = 0; nf < 2; ++nf)
            #pragma unroll
            for (int rr = 0; rr < 8; ++rr)
                sf[nf][rr] = __expf(sf[nf][rr] - mrow[rr]);
        #pragma unroll
        for (int rr = 0; rr < 8; ++rr) ps[rr] = sf[0][rr] + sf[1][rr];
        #pragma unroll
        for (int m = 1; m <= 8; m <<= 1)
            #pragma unroll
            for (int rr = 0; rr < 8; ++rr) ps[rr] += __shfl_xor(ps[rr], m, 32);
        #pragma unroll
        for (int rr = 0; rr < 8; ++rr) lrow[rr] = lrow[rr] * scl[rr] + ps[rr];
        #pragma unroll
        for (int nf = 0; nf < 4; ++nf)
            #pragma unroll
            for (int rr = 0; rr < 8; ++rr) acc[nf][rr] *= scl[rr];

        // stage P (bf16) to wave-local LDS in row-major, reload as A-fragment
        bf16_t* pp = Ps[wid];
        #pragma unroll
        for (int nf = 0; nf < 2; ++nf)
            #pragma unroll
            for (int rr = 0; rr < 8; ++rr)
                pp[((lane >> 4) * 8 + rr) * LVT + nf * 16 + fm] = (bf16_t)sf[nf][rr];
        v16bf pa = frag_a(&pp[fm * LVT + fh8]);

        // acc += P @ V  (V split hi/lo)
        #pragma unroll
        for (int nf = 0; nf < 4; ++nf) {
            v16bf bvh = frag_b(&Vth[(nf * 16 + fm) * LVT + fh8]);
            v16bf bvl = frag_b(&Vtl[(nf * 16 + fm) * LVT + fh8]);
            acc[nf] = wmma_bf16(pa, bvh, acc[nf]);
            acc[nf] = wmma_bf16(pa, bvl, acc[nf]);
        }
    }

    #pragma unroll
    for (int nf = 0; nf < 4; ++nf)
        #pragma unroll
        for (int rr = 0; rr < 8; ++rr) {
            int row = qbase + (lane >> 4) * 8 + rr;
            int col = head * HEAD_DIM + nf * 16 + fm;
            O[(size_t)row * D_MODEL + col] = acc[nf][rr] / lrow[rr];
        }
}

// ---------------------------------------------------------------------------
// Host orchestration
// ---------------------------------------------------------------------------
extern "C" void kernel_launch(void* const* d_in, const int* in_sizes, int n_in,
                              void* d_out, int out_size, void* d_ws, size_t ws_size,
                              hipStream_t stream) {
    (void)in_sizes; (void)n_in; (void)out_size; (void)ws_size;

    const float* x   = (const float*)d_in[0];
    // d_in[1] = start_pos (0), d_in[2] = mask (causal; handled analytically)
    const float* cs  = (const float*)d_in[3];
    const float* sn  = (const float*)d_in[4];
    const float* wq  = (const float*)d_in[5];
    const float* wk  = (const float*)d_in[6];
    const float* wv  = (const float*)d_in[7];
    const float* wo  = (const float*)d_in[8];
    const float* lqr = (const float*)d_in[9];
    const float* lqA = (const float*)d_in[10];
    const float* lqB = (const float*)d_in[11];
    const float* lkr = (const float*)d_in[12];
    const float* lkA = (const float*)d_in[13];
    const float* lkB = (const float*)d_in[14];
    const float* lvr = (const float*)d_in[15];
    const float* lvA = (const float*)d_in[16];
    const float* lvB = (const float*)d_in[17];
    const float* lor = (const float*)d_in[18];
    const float* loA = (const float*)d_in[19];
    const float* loB = (const float*)d_in[20];
    float* out = (float*)d_out;

    char* ws = (char*)d_ws;
    size_t off = 0;
    auto alloc = [&](size_t bytes) -> void* {
        void* p = ws + off;
        off += (bytes + 255) & ~(size_t)255;
        return p;
    };

    const size_t SD = (size_t)S_LEN * D_MODEL;   // 4M elems
    const size_t SN = (size_t)S_LEN * NKV;       // 1M elems
    const size_t DD = (size_t)D_MODEL * D_MODEL; // wq/wo elems
    const size_t ND = (size_t)NKV * D_MODEL;     // wk/wv elems

    bf16_t* xh  = (bf16_t*)alloc(2 * SD);
    bf16_t* xl  = (bf16_t*)alloc(2 * SD);
    bf16_t* wqh = (bf16_t*)alloc(2 * DD);
    bf16_t* wql = (bf16_t*)alloc(2 * DD);
    bf16_t* wkh = (bf16_t*)alloc(2 * ND);
    bf16_t* wkl = (bf16_t*)alloc(2 * ND);
    bf16_t* wvh = (bf16_t*)alloc(2 * ND);
    bf16_t* wvl = (bf16_t*)alloc(2 * ND);
    bf16_t* woh = (bf16_t*)alloc(2 * DD);
    bf16_t* wol = (bf16_t*)alloc(2 * DD);
    float* loraq = (float*)alloc(4 * SD);
    float* lorak = (float*)alloc(4 * SN);
    float* lorav = (float*)alloc(4 * SN);
    float* lorao = (float*)alloc(4 * SD);
    float* xq = (float*)alloc(4 * SD);
    float* xk = (float*)alloc(4 * SN);
    float* xv = (float*)alloc(4 * SN);
    bf16_t* qhB = (bf16_t*)alloc(2 * SD);
    bf16_t* qlB = (bf16_t*)alloc(2 * SD);
    bf16_t* khB = (bf16_t*)alloc(2 * SN);
    bf16_t* klB = (bf16_t*)alloc(2 * SN);
    bf16_t* vhB = (bf16_t*)alloc(2 * SN);
    bf16_t* vlB = (bf16_t*)alloc(2 * SN);
    float* ao = (float*)alloc(4 * SD);
    bf16_t* aoh = (bf16_t*)alloc(2 * SD);
    bf16_t* aol = (bf16_t*)alloc(2 * SD);

    // 1) split inputs/weights into bf16 hi/lo
    split_bf16_kernel<<<2048, 256, 0, stream>>>(x,  xh,  xl,  (int)SD);
    split_bf16_kernel<<<2048, 256, 0, stream>>>(wq, wqh, wql, (int)DD);
    split_bf16_kernel<<<2048, 256, 0, stream>>>(wk, wkh, wkl, (int)ND);
    split_bf16_kernel<<<2048, 256, 0, stream>>>(wv, wvh, wvl, (int)ND);
    split_bf16_kernel<<<2048, 256, 0, stream>>>(wo, woh, wol, (int)DD);

    // 2) MoE-LoRA biases for q/k/v
    lora_kernel<<<S_LEN, 256, 0, stream>>>(x, lqr, lqA, lqB, loraq, D_MODEL);
    lora_kernel<<<S_LEN, 256, 0, stream>>>(x, lkr, lkA, lkB, lorak, NKV);
    lora_kernel<<<S_LEN, 256, 0, stream>>>(x, lvr, lvA, lvB, lorav, NKV);

    // 3) projections (split-bf16 WMMA GEMM + LoRA bias)
    gemm_split_kernel<<<dim3(D_MODEL / 128, S_LEN / 128), 256, 0, stream>>>(
        xh, xl, wqh, wql, loraq, xq, D_MODEL, D_MODEL);
    gemm_split_kernel<<<dim3(NKV / 128, S_LEN / 128), 256, 0, stream>>>(
        xh, xl, wkh, wkl, lorak, xk, NKV, D_MODEL);
    gemm_split_kernel<<<dim3(NKV / 128, S_LEN / 128), 256, 0, stream>>>(
        xh, xl, wvh, wvl, lorav, xv, NKV, D_MODEL);

    // 4) RoPE on q and k
    {
        int tq = S_LEN * N_HEADS * (HEAD_DIM / 2);
        rope_kernel<<<(tq + 255) / 256, 256, 0, stream>>>(xq, cs, sn, N_HEADS);
        int tk = S_LEN * N_KVH * (HEAD_DIM / 2);
        rope_kernel<<<(tk + 255) / 256, 256, 0, stream>>>(xk, cs, sn, N_KVH);
    }

    // 5) split q/k/v to bf16 hi/lo
    split_bf16_kernel<<<2048, 256, 0, stream>>>(xq, qhB, qlB, (int)SD);
    split_bf16_kernel<<<2048, 256, 0, stream>>>(xk, khB, klB, (int)SN);
    split_bf16_kernel<<<2048, 256, 0, stream>>>(xv, vhB, vlB, (int)SN);

    // 6) causal flash attention
    attn_kernel<<<dim3(S_LEN / 128, N_HEADS), 256, 0, stream>>>(
        qhB, qlB, khB, klB, vhB, vlB, ao);

    // 7) output LoRA + final projection into d_out
    lora_kernel<<<S_LEN, 256, 0, stream>>>(ao, lor, loA, loB, lorao, D_MODEL);
    split_bf16_kernel<<<2048, 256, 0, stream>>>(ao, aoh, aol, (int)SD);
    gemm_split_kernel<<<dim3(D_MODEL / 128, S_LEN / 128), 256, 0, stream>>>(
        aoh, aol, woh, wol, lorao, out, D_MODEL, D_MODEL);
}
